// TransformerEncoderV2_36275293782329
// MI455X (gfx1250) — matrix operand order repaired and checked by reading, not compiled, not run on hardware
//
#include <hip/hip_runtime.h>
#include <hip/hip_bf16.h>
#include <math.h>
#include <stdint.h>

// ---------------------------------------------------------------------------
// Types & helpers
// ---------------------------------------------------------------------------
typedef __bf16 bf16_t;
typedef __attribute__((ext_vector_type(16))) __bf16 v16bf;
typedef __attribute__((ext_vector_type(8)))  __bf16 v8bf;   // 16 bytes
typedef __attribute__((ext_vector_type(8)))  float  v8f;

static __device__ __forceinline__ v16bf join8(v8bf lo, v8bf hi) {
  return __builtin_shufflevector(lo, hi, 0,1,2,3,4,5,6,7,8,9,10,11,12,13,14,15);
}
static __device__ __forceinline__ v8f wmma_bf16(v16bf a, v16bf b, v8f c) {
  return __builtin_amdgcn_wmma_f32_16x16x32_bf16(false, a, false, b, (short)0, c,
                                                 false, false);
}

// Async global->LDS 16B copy per lane (gfx1250 ASYNC path, tracked by ASYNCcnt).
static __device__ __forceinline__ void async_load_b128(const void* gptr, void* lptr) {
  unsigned lds = (unsigned)(uintptr_t)lptr;               // low 32b = LDS address
  unsigned long long g = (unsigned long long)(uintptr_t)gptr;
  asm volatile("global_load_async_to_lds_b128 %0, %1, off"
               :: "v"(lds), "v"(g) : "memory");
}
static __device__ __forceinline__ void wait_async0() {
  asm volatile("s_wait_asynccnt 0" ::: "memory");
}

// VALU-only butterfly max across a 16-lane row half (no LDS pipe usage).
template <int CTRL>
static __device__ __forceinline__ float dpp_movf(float x) {
  int r = __builtin_amdgcn_mov_dpp(__builtin_bit_cast(int, x), CTRL, 0xF, 0xF, true);
  return __builtin_bit_cast(float, r);
}
static __device__ __forceinline__ float rowmax16(float x) {
  x = fmaxf(x, dpp_movf<0x0B1>(x));   // quad_perm [1,0,3,2]  (xor 1)
  x = fmaxf(x, dpp_movf<0x04E>(x));   // quad_perm [2,3,0,1]  (xor 2)
  x = fmaxf(x, dpp_movf<0x141>(x));   // row_half_mirror      (fold quads in 8)
  x = fmaxf(x, dpp_movf<0x140>(x));   // row_mirror           (fold halves in 16)
  return x;
}

#define EMB   768
#define SEQ   2048
#define BATCH 4
#define HEADS 12
#define HDIM  64
#define MTOT  (BATCH * SEQ)   // 8192 tokens

// ---------------------------------------------------------------------------
// fp32 [K][N] -> bf16 transposed [N][K]  (one-time; counts divisible by 256)
// ---------------------------------------------------------------------------
__global__ __launch_bounds__(256)
void cvt_transpose_kernel(const float* __restrict__ in, bf16_t* __restrict__ out,
                          int K, int N) {
  size_t i = (size_t)blockIdx.x * 256 + threadIdx.x;  // over K*N, coalesced read
  int k = (int)(i / N), n = (int)(i % N);
  out[(size_t)n * K + k] = (bf16_t)in[i];
}

// ---------------------------------------------------------------------------
// LayerNorm over 768 columns; one block per row; 256 threads (3 cols each).
// ---------------------------------------------------------------------------
template <bool WRITE_F32>
__global__ __launch_bounds__(256)
void layernorm_kernel(const float* __restrict__ x, const float* __restrict__ w,
                      const float* __restrict__ b, bf16_t* __restrict__ outB,
                      float* __restrict__ outF) {
  __shared__ float red[8];
  const int row  = blockIdx.x;
  const int tid  = threadIdx.x;
  const int wave = tid >> 5, lane = tid & 31;
  const float* xr = x + (size_t)row * EMB;

  float v[3];
  float s = 0.f;
#pragma unroll
  for (int i = 0; i < 3; i++) { v[i] = xr[tid + i * 256]; s += v[i]; }
#pragma unroll
  for (int off = 16; off > 0; off >>= 1) s += __shfl_xor(s, off);
  if (lane == 0) red[wave] = s;
  __syncthreads();
  if (tid == 0) {
    float t = 0.f;
#pragma unroll
    for (int i = 0; i < 8; i++) t += red[i];
    red[0] = t * (1.0f / EMB);
  }
  __syncthreads();
  const float mu = red[0];
  __syncthreads();

  float q = 0.f;
#pragma unroll
  for (int i = 0; i < 3; i++) { float d = v[i] - mu; q += d * d; }
#pragma unroll
  for (int off = 16; off > 0; off >>= 1) q += __shfl_xor(q, off);
  if (lane == 0) red[wave] = q;
  __syncthreads();
  if (tid == 0) {
    float t = 0.f;
#pragma unroll
    for (int i = 0; i < 8; i++) t += red[i];
    red[0] = rsqrtf(t * (1.0f / EMB) + 1e-5f);
  }
  __syncthreads();
  const float rstd = red[0];

#pragma unroll
  for (int i = 0; i < 3; i++) {
    int c = tid + i * 256;
    float y = (v[i] - mu) * rstd * w[c] + b[c];
    outB[(size_t)row * EMB + c] = (bf16_t)y;
    if (WRITE_F32) outF[(size_t)row * EMB + c] = y;
  }
}

// ---------------------------------------------------------------------------
// bf16 WMMA GEMM: C[M,N] = epilogue(A[M,K] @ W[K,N]); W passed TRANSPOSED [N][K].
// Tile 128x64, BK=32; 256 threads = 8 waves, each wave 32x32.
// Both tiles streamed with async-to-LDS (double-buffered, ASYNCcnt).
// ---------------------------------------------------------------------------
#define BM 128
#define BN 64
#define BK 32
#define LDA_S 40   // padded LDS row stride (bf16 elements)
#define LDB_S 40

enum { EPI_QKV = 0, EPI_BIAS_RES_F32 = 1, EPI_BIAS_GELU_B16 = 2 };

template <int EPI>
__global__ __launch_bounds__(256)
void gemm_bf16_wmma(const bf16_t* __restrict__ A, const bf16_t* __restrict__ Wt,
                    const float* __restrict__ bias, const float* __restrict__ resid,
                    float* __restrict__ outF, bf16_t* __restrict__ outB,
                    bf16_t* __restrict__ qkOut, bf16_t* __restrict__ vtOut,
                    int M, int N, int K) {
  __shared__ bf16_t lA[2][BM * LDA_S];
  __shared__ bf16_t lW[2][BN * LDB_S];

  const int tid  = threadIdx.x;
  const int wave = tid >> 5, lane = tid & 31;
  const int lm = lane & 15, kh = lane >> 4;
  const int m0 = blockIdx.y * BM, n0 = blockIdx.x * BN;
  const int wm = (wave >> 1) * 32;
  const int wn = (wave & 1) * 32;

  // Per-thread staging assignment (all 16B chunks, K-contiguous in both tiles).
  const int ar0 = tid >> 2, akc = (tid & 3) * 8;     // A rows tid>>2 and +64
  const int wr0 = tid >> 2, wkc = (tid & 3) * 8;     // W rows 0..63

  v8f acc[2][2] = {};

  // Prologue: stream tile k0=0 into buffer 0.
  {
    async_load_b128(A + (size_t)(m0 + ar0) * K + akc, &lA[0][ar0 * LDA_S + akc]);
    async_load_b128(A + (size_t)(m0 + ar0 + 64) * K + akc,
                    &lA[0][(ar0 + 64) * LDA_S + akc]);
    async_load_b128(Wt + (size_t)(n0 + wr0) * K + wkc, &lW[0][wr0 * LDB_S + wkc]);
  }

  int buf = 0;
  for (int k0 = 0; k0 < K; k0 += BK) {
    wait_async0();        // my async copies into lA[buf]/lW[buf] landed
    __syncthreads();      // everyone's did

    if (k0 + BK < K) {    // stream next tile into the other buffer
      int kn = k0 + BK;
      async_load_b128(A + (size_t)(m0 + ar0) * K + kn + akc,
                      &lA[buf ^ 1][ar0 * LDA_S + akc]);
      async_load_b128(A + (size_t)(m0 + ar0 + 64) * K + kn + akc,
                      &lA[buf ^ 1][(ar0 + 64) * LDA_S + akc]);
      async_load_b128(Wt + (size_t)(n0 + wr0) * K + kn + wkc,
                      &lW[buf ^ 1][wr0 * LDB_S + wkc]);
    }

    v16bf af[2], bfr[2];
#pragma unroll
    for (int ms = 0; ms < 2; ms++) {
      const bf16_t* p = &lA[buf][(wm + ms * 16 + lm) * LDA_S];
      af[ms] = join8(*(const v8bf*)(p + kh * 8), *(const v8bf*)(p + 16 + kh * 8));
    }
#pragma unroll
    for (int ns = 0; ns < 2; ns++) {
      const bf16_t* p = &lW[buf][(wn + ns * 16 + lm) * LDB_S + kh * 8];
      bfr[ns] = join8(*(const v8bf*)(p), *(const v8bf*)(p + 16));
    }
#pragma unroll
    for (int ms = 0; ms < 2; ms++)
#pragma unroll
      for (int ns = 0; ns < 2; ns++)
        acc[ms][ns] = wmma_bf16(af[ms], bfr[ns], acc[ms][ns]);

    buf ^= 1;
  }

  // Epilogue. C layout: lanes 0-15 -> M=v, N=lane; lanes 16-31 -> M=v+8.
#pragma unroll
  for (int ms = 0; ms < 2; ms++) {
#pragma unroll
    for (int ns = 0; ns < 2; ns++) {
#pragma unroll
      for (int v = 0; v < 8; v++) {
        int row = m0 + wm + ms * 16 + v + kh * 8;
        int col = n0 + wn + ns * 16 + lm;
        float val = acc[ms][ns][v];
        size_t idx = (size_t)row * N + col;
        if (EPI == EPI_QKV) {
          if (col < 2 * EMB) {
            qkOut[(size_t)row * (2 * EMB) + col] = (bf16_t)val;
          } else {
            int c = col - 2 * EMB, h = c >> 6, d = c & 63;
            int bb = row >> 11, sq = row & (SEQ - 1);
            vtOut[(((size_t)(bb * HEADS + h) * HDIM + d) << 11) + sq] = (bf16_t)val;
          }
        } else if (EPI == EPI_BIAS_RES_F32) {
          outF[idx] = val + bias[col] + resid[idx];
        } else {  // EPI_BIAS_GELU_B16 : exact GELU
          float t = val + bias[col];
          float g = 0.5f * t * (1.0f + erff(t * 0.70710678118654752f));
          outB[idx] = (bf16_t)g;
        }
      }
    }
  }
}

// ---------------------------------------------------------------------------
// Flash attention (non-causal), bf16 WMMA, fp32 online softmax.
// grid = (SEQ/128, BATCH*HEADS); 8 waves; each wave owns 16 q-rows.
// Row-max: DPP butterfly (VALU only). Row-sum: P @ ones via an extra WMMA.
// ---------------------------------------------------------------------------
__global__ __launch_bounds__(256)
void attention_kernel(const bf16_t* __restrict__ qk, const bf16_t* __restrict__ vt,
                      bf16_t* __restrict__ attn) {
  __shared__ bf16_t pbuf[8][16 * 40];   // per-wave P transpose buffer

  const int tid  = threadIdx.x;
  const int wave = tid >> 5, lane = tid & 31;
  const int lm = lane & 15, kh = lane >> 4;
  const int bh = blockIdx.y;
  const int b = bh / HEADS, h = bh % HEADS;
  const size_t tq = (size_t)b * SEQ + blockIdx.x * 128 + wave * 16;

  const bf16_t* Qp = qk + tq * (2 * EMB) + h * HDIM;
  const bf16_t* Kp = qk + (size_t)b * SEQ * (2 * EMB) + EMB + h * HDIM;
  const bf16_t* Vp = vt + (size_t)bh * HDIM * SEQ;
  bf16_t* pb = pbuf[wave];

  // Q fragments, pre-scaled by 1/sqrt(64) = 0.125 (power of two: exact in bf16).
  v16bf aq[2];
#pragma unroll
  for (int s = 0; s < 2; s++) {
    const bf16_t* qr = Qp + (size_t)lm * (2 * EMB) + s * 32 + kh * 8;
    v16bf q = join8(*(const v8bf*)(qr), *(const v8bf*)(qr + 16));
#pragma unroll
    for (int e = 0; e < 16; e++) q[e] = q[e] * (__bf16)0.125f;
    aq[s] = q;
  }

  v16bf ones;
#pragma unroll
  for (int e = 0; e < 16; e++) ones[e] = (__bf16)1.0f;

  v8f o[4] = {};
  v8f ol = {};              // online-softmax denominator, via ones-WMMA
  float mstat[8];
#pragma unroll
  for (int v = 0; v < 8; v++) mstat[v] = -1e30f;

  for (int kk = 0; kk < SEQ; kk += 32) {
    // S = (Q*scale) @ K^T : 16x32 strip = two 16-key WMMA tiles.
    v8f sc[2] = {};
#pragma unroll
    for (int nt = 0; nt < 2; nt++) {
#pragma unroll
      for (int s = 0; s < 2; s++) {
        const bf16_t* kr =
            Kp + (size_t)(kk + nt * 16 + lm) * (2 * EMB) + s * 32 + kh * 8;
        v16bf bk = join8(*(const v8bf*)(kr), *(const v8bf*)(kr + 16));
        sc[nt] = wmma_bf16(aq[s], bk, sc[nt]);
      }
    }

    // Online softmax; each C row lives in one 16-lane half.
#pragma unroll
    for (int v = 0; v < 8; v++) {
      float s0 = sc[0][v], s1 = sc[1][v];
      float rm = rowmax16(fmaxf(s0, s1));
      float mnew = fmaxf(mstat[v], rm);
      float corr = __expf(mstat[v] - mnew);
      mstat[v] = mnew;
      float p0 = __expf(s0 - mnew), p1 = __expf(s1 - mnew);
#pragma unroll
      for (int dt = 0; dt < 4; dt++) o[dt][v] *= corr;
      ol[v] *= corr;
      int prow = v + kh * 8;
      pb[prow * 40 + lm]      = (bf16_t)p0;
      pb[prow * 40 + 16 + lm] = (bf16_t)p1;
    }
    // Wave-private LDS transpose: wait for our own stores before cross-lane reads.
    asm volatile("s_wait_dscnt 0" ::: "memory");

    const bf16_t* pr = pb + lm * 40 + kh * 8;
    v16bf pf = join8(*(const v8bf*)(pr), *(const v8bf*)(pr + 16));

    // O += P @ V (V^T rows: keys contiguous); l += P @ 1 on the matrix unit.
#pragma unroll
    for (int dt = 0; dt < 4; dt++) {
      const bf16_t* vr = Vp + (size_t)(dt * 16 + lm) * SEQ + kk + kh * 8;
      v16bf bv = join8(*(const v8bf*)(vr), *(const v8bf*)(vr + 16));
      o[dt] = wmma_bf16(pf, bv, o[dt]);
    }
    ol = wmma_bf16(pf, ones, ol);
  }

  // Normalize and store token-major bf16.
#pragma unroll
  for (int v = 0; v < 8; v++) {
    float inv = 1.0f / ol[v];
    size_t t = tq + v + kh * 8;
#pragma unroll
    for (int dt = 0; dt < 4; dt++)
      attn[t * EMB + h * HDIM + dt * 16 + lm] = (bf16_t)(o[dt][v] * inv);
  }
}

// ---------------------------------------------------------------------------
// Host-side orchestration
// ---------------------------------------------------------------------------
extern "C" void kernel_launch(void* const* d_in, const int* in_sizes, int n_in,
                              void* d_out, int out_size, void* d_ws, size_t ws_size,
                              hipStream_t stream) {
  const float* x     = (const float*)d_in[0];
  const float* w_qkv = (const float*)d_in[1];
  const float* w_o   = (const float*)d_in[2];
  const float* b_o   = (const float*)d_in[3];
  const float* ln1w  = (const float*)d_in[4];
  const float* ln1b  = (const float*)d_in[5];
  const float* ln2w  = (const float*)d_in[6];
  const float* ln2b  = (const float*)d_in[7];
  const float* w1    = (const float*)d_in[8];
  const float* b1    = (const float*)d_in[9];
  const float* w2    = (const float*)d_in[10];
  const float* b2    = (const float*)d_in[11];
  float* out = (float*)d_out;

  char* ws = (char*)d_ws;
  size_t off = 0;
  auto alloc = [&](size_t bytes) -> void* {
    off = (off + 255) & ~(size_t)255;
    void* p = ws + off;
    off += bytes;
    return p;
  };

  // Region plan (lifetime-based reuse):
  bf16_t* X1   = (bf16_t*)alloc((size_t)MTOT * EMB * 2);         // LN1 out -> attn
  bf16_t* qkb  = (bf16_t*)alloc((size_t)MTOT * 2 * EMB * 2);     // Q|K -> resid f32 -> H
  bf16_t* vtb  = (bf16_t*)alloc((size_t)BATCH * HEADS * HDIM * SEQ * 2); // V^T -> X2
  float*  X2f  = (float*)alloc((size_t)MTOT * EMB * 4);          // LN2 fp32 residual
  bf16_t* wqkvT = (bf16_t*)alloc((size_t)EMB * 3 * EMB * 2);     // transposed bf16 weights
  bf16_t* woT   = (bf16_t*)alloc((size_t)EMB * EMB * 2);
  bf16_t* w1T   = (bf16_t*)alloc((size_t)EMB * 2 * EMB * 2);
  bf16_t* w2T   = (bf16_t*)alloc((size_t)2 * EMB * EMB * 2);

  bf16_t* attnB = X1;            // X1 dead after QKV GEMM
  float*  resid = (float*)qkb;   // Q|K dead after attention (M*1536*2 == M*768*4)
  bf16_t* X2    = vtb;           // V^T dead after attention
  bf16_t* H     = (bf16_t*)qkb;  // resid dead after LN2

  // 1) Weights fp32 -> bf16, transposed to [N][K].
  cvt_transpose_kernel<<<EMB * 3 * EMB / 256, 256, 0, stream>>>(w_qkv, wqkvT, EMB, 3 * EMB);
  cvt_transpose_kernel<<<EMB * EMB / 256, 256, 0, stream>>>(w_o, woT, EMB, EMB);
  cvt_transpose_kernel<<<EMB * 2 * EMB / 256, 256, 0, stream>>>(w1, w1T, EMB, 2 * EMB);
  cvt_transpose_kernel<<<2 * EMB * EMB / 256, 256, 0, stream>>>(w2, w2T, 2 * EMB, EMB);

  // 2) LN1 -> X1 (bf16).
  layernorm_kernel<false><<<MTOT, 256, 0, stream>>>(x, ln1w, ln1b, X1, nullptr);

  // 3) QKV GEMM: X1 @ w_qkv -> Q|K token-major bf16, V transposed bf16.
  gemm_bf16_wmma<EPI_QKV><<<dim3(3 * EMB / BN, MTOT / BM), 256, 0, stream>>>(
      X1, wqkvT, nullptr, nullptr, nullptr, nullptr, qkb, vtb, MTOT, 3 * EMB, EMB);

  // 4) Flash attention -> attnB (bf16, token-major).
  attention_kernel<<<dim3(SEQ / 128, BATCH * HEADS), 256, 0, stream>>>(qkb, vtb, attnB);

  // 5) Output projection + residual: resid = attn @ w_o + b_o + x (fp32).
  gemm_bf16_wmma<EPI_BIAS_RES_F32><<<dim3(EMB / BN, MTOT / BM), 256, 0, stream>>>(
      attnB, woT, b_o, x, resid, nullptr, nullptr, nullptr, MTOT, EMB, EMB);

  // 6) LN2 -> X2 (bf16) and X2f (fp32).
  layernorm_kernel<true><<<MTOT, 256, 0, stream>>>(resid, ln2w, ln2b, X2, X2f);

  // 7) MLP up + exact GELU: H = gelu(X2 @ w1 + b1) (bf16).
  gemm_bf16_wmma<EPI_BIAS_GELU_B16><<<dim3(2 * EMB / BN, MTOT / BM), 256, 0, stream>>>(
      X2, w1T, b1, nullptr, nullptr, H, nullptr, nullptr, MTOT, 2 * EMB, EMB);

  // 8) MLP down + residual: out = X2f + (H @ w2 + b2) (fp32).
  gemm_bf16_wmma<EPI_BIAS_RES_F32><<<dim3(EMB / BN, MTOT / BM), 256, 0, stream>>>(
      H, w2T, b2, X2f, out, nullptr, nullptr, nullptr, MTOT, EMB, 2 * EMB);

  (void)in_sizes; (void)n_in; (void)out_size; (void)ws_size;
}
